// GNNBind_22230750724684
// MI455X (gfx1250) — matrix operand order repaired and checked by reading, not compiled, not run on hardware
//
#include <hip/hip_runtime.h>

typedef __bf16 bf16;
typedef __attribute__((ext_vector_type(16))) __bf16 v16bf;
typedef __attribute__((ext_vector_type(8)))  float  v8f;

struct __attribute__((aligned(16))) U4 { unsigned int x, y, z, w; };

union FragAB { v16bf v; U4 q[2]; bf16 h[16]; };
union FragC  { v8f  v; float f[8]; };

enum { NLIG = 768, NREC = 4096, ELIG = 1536, EREC = 16384 };

__device__ __forceinline__ v8f wmma_bf(v16bf a, v16bf b, v8f c) {
  // D = A(16x32 bf16) * B(32x16 bf16) + C(16x16 f32)
  return __builtin_amdgcn_wmma_f32_16x16x32_bf16(false, a, false, b, (short)0, c, false, false);
}

// ---------------- generic bf16 WMMA GEMM:  out = act((A @ Wt^T + bias) * scale) ----------------
// A: [M x K] bf16 row-major (lda in elems).  Wt: [N x K] bf16 (pre-transposed weights).
__global__ __launch_bounds__(128) void k_gemm(
    const bf16* __restrict__ A, int lda,
    const bf16* __restrict__ Wt, int K,
    const float* __restrict__ bias,
    float* __restrict__ outF, int ldF,
    bf16* __restrict__ outB, int ldB, int outT,
    int M, int N, int act, float scale)
{
  int lane = threadIdx.x & 31, wave = threadIdx.x >> 5;
  int m0 = (blockIdx.y * 4 + wave) * 16;
  if (m0 >= M) return;
  int n0 = blockIdx.x * 16;
  int m = lane & 15, hi = lane >> 4;
  const bf16* arow = A + (size_t)(m0 + m) * lda;
  const bf16* brow = Wt + (size_t)(n0 + m) * K;
  FragC c;
#pragma unroll
  for (int i = 0; i < 8; ++i) c.f[i] = 0.f;
  for (int k0 = 0; k0 < K; k0 += 32) {
    FragAB a, b;
    a.q[0] = *(const U4*)(arow + k0 + hi * 8);        // halfs 0..7  : K = k0 + hi*8 + j
    a.q[1] = *(const U4*)(arow + k0 + 16 + hi * 8);   // halfs 8..15 : K = k0+16+hi*8+j
    b.q[0] = *(const U4*)(brow + k0 + hi * 8);        // K = k0 + hi*8 + 0..7
    b.q[1] = *(const U4*)(brow + k0 + 16 + hi * 8);   // K = k0 + 16 + hi*8 + 0..7
    __builtin_prefetch(arow + k0 + 64, 0, 1);
    __builtin_prefetch(brow + k0 + 64, 0, 1);
    c.v = wmma_bf(a.v, b.v, c.v);
  }
#pragma unroll
  for (int r = 0; r < 8; ++r) {
    int mm = m0 + r + hi * 8;
    int nn = n0 + m;
    float v = c.f[r];
    if (bias) v += bias[nn];
    v *= scale;
    if (act == 1) v = fmaxf(v, 0.f);
    if (outF) outF[(size_t)mm * ldF + nn] = v;
    if (outB) {
      if (outT) outB[(size_t)nn * ldB + mm] = (bf16)v;
      else      outB[(size_t)mm * ldB + nn] = (bf16)v;
    }
  }
}

// -------- fused NNConv message: msg[e,:] += (eh[e] (x) hid[src[e]]) @ W2r ;  scatter-add to dst --------
// W2rT: [64 x 4160] bf16, row o holds k=(a*64+b): a<64 -> e2w[a, b*64+o], a==64 -> e2b[b*64+o]
// Per 32-K chunk the eh index a=k>>6 is constant and the hid index b=k&63 sweeps exactly the
// A-fragment pattern of a 32-half hid range, so A-frag = eh_scalar * preloaded hid fragment.
__global__ __launch_bounds__(128) void k_edge_msg(
    const bf16* __restrict__ eh, const bf16* __restrict__ hid,
    const int* __restrict__ src, const int* __restrict__ dst,
    const bf16* __restrict__ W2rT, float* __restrict__ xbuf)
{
  __shared__ bf16 s_hid[64][64];
  __shared__ bf16 s_eh[64][64];
  int e0 = blockIdx.x * 64;
  int t = threadIdx.x;
  {
    int row = t >> 1, half = (t & 1) * 32;
    int e = e0 + row;
    int s = src[e];
    const U4* ph = (const U4*)(hid + (size_t)s * 64 + half);
    U4* dh = (U4*)&s_hid[row][half];
    dh[0] = ph[0]; dh[1] = ph[1]; dh[2] = ph[2]; dh[3] = ph[3];
    const U4* pe = (const U4*)(eh + (size_t)e * 64 + half);
    U4* de = (U4*)&s_eh[row][half];
    de[0] = pe[0]; de[1] = pe[1]; de[2] = pe[2]; de[3] = pe[3];
  }
  __syncthreads();
  int lane = t & 31, wave = t >> 5;
  int m = lane & 15, hi = lane >> 4;
  int er = wave * 16 + m;
  // preload the two hid A-fragments (b-range 0..31 and 32..63) once
  FragAB hf[2];
  hf[0].q[0] = *(const U4*)&s_hid[er][hi * 8];
  hf[0].q[1] = *(const U4*)&s_hid[er][16 + hi * 8];
  hf[1].q[0] = *(const U4*)&s_hid[er][32 + hi * 8];
  hf[1].q[1] = *(const U4*)&s_hid[er][48 + hi * 8];
  const bf16* bbase = W2rT + (size_t)m * 4160 + hi * 8;
  FragC c[4];
#pragma unroll
  for (int nt = 0; nt < 4; ++nt)
#pragma unroll
    for (int i = 0; i < 8; ++i) c[nt].f[i] = 0.f;
  for (int a = 0; a < 65; ++a) {
    float ev = (a == 64) ? 1.f : (float)s_eh[er][a];
#pragma unroll
    for (int half = 0; half < 2; ++half) {
      int k0 = a * 64 + half * 32;
      FragAB af;
#pragma unroll
      for (int j = 0; j < 16; ++j)
        af.h[j] = (bf16)(ev * (float)hf[half].h[j]);
#pragma unroll
      for (int nt = 0; nt < 4; ++nt) {
        FragAB b;
        const bf16* brow = bbase + (size_t)nt * 16 * 4160 + k0;
        b.q[0] = *(const U4*)(brow);
        b.q[1] = *(const U4*)(brow + 16);
        c[nt].v = wmma_bf(af.v, b.v, c[nt].v);
      }
    }
  }
#pragma unroll
  for (int nt = 0; nt < 4; ++nt)
#pragma unroll
    for (int r = 0; r < 8; ++r) {
      int el = wave * 16 + r + hi * 8;
      int d = dst[e0 + el];
      atomicAdd(&xbuf[(size_t)d * 64 + nt * 16 + m], c[nt].f[r]);
    }
}

// -------- per-graph cross attention: logits=q@k^T (WMMA), softmax (VALU+shfl), P@V (WMMA) --------
__global__ __launch_bounds__(96) void k_attn(
    const bf16* __restrict__ qb, const bf16* __restrict__ kb,
    const bf16* __restrict__ vtb, bf16* __restrict__ atn)
{
  __shared__ float s_l[48][256];
  __shared__ bf16  s_p[48][256];
  int g = blockIdx.x;
  int t = threadIdx.x, lane = t & 31, wave = t >> 5;
  int m = lane & 15, hi = lane >> 4;
  const bf16* qrow = qb + ((size_t)g * 48 + wave * 16 + m) * 64;
  for (int nt = 0; nt < 16; ++nt) {
    FragC c;
#pragma unroll
    for (int i = 0; i < 8; ++i) c.f[i] = 0.f;
#pragma unroll
    for (int k0 = 0; k0 < 64; k0 += 32) {
      FragAB a, b;
      a.q[0] = *(const U4*)(qrow + k0 + hi * 8);
      a.q[1] = *(const U4*)(qrow + k0 + 16 + hi * 8);
      const bf16* krow = kb + ((size_t)g * 256 + nt * 16 + m) * 64 + k0 + hi * 8;
      b.q[0] = *(const U4*)(krow);
      b.q[1] = *(const U4*)(krow + 16);
      c.v = wmma_bf(a.v, b.v, c.v);
    }
#pragma unroll
    for (int r = 0; r < 8; ++r)
      s_l[wave * 16 + r + hi * 8][nt * 16 + m] = c.f[r];
  }
  __syncthreads();
  {
    // lane pair (L, L+16) splits row's 256 cols; combine partials via wave32 shfl_xor
    int row = wave * 16 + m;
    int c0 = hi * 128;
    float mx = -3.4e38f;
    for (int ci = 0; ci < 128; ++ci) mx = fmaxf(mx, s_l[row][c0 + ci]);
    mx = fmaxf(mx, __shfl_xor(mx, 16, 32));
    float s = 0.f;
    for (int ci = 0; ci < 128; ++ci) {
      float e = __expf(s_l[row][c0 + ci] - mx);
      s_l[row][c0 + ci] = e;
      s += e;
    }
    s += __shfl_xor(s, 16, 32);
    float inv = 1.f / s;
    for (int ci = 0; ci < 128; ++ci)
      s_p[row][c0 + ci] = (bf16)(s_l[row][c0 + ci] * inv);
  }
  __syncthreads();
  FragC c2[4];
#pragma unroll
  for (int nt = 0; nt < 4; ++nt)
#pragma unroll
    for (int i = 0; i < 8; ++i) c2[nt].f[i] = 0.f;
  for (int k0 = 0; k0 < 256; k0 += 32) {
    FragAB a;
    a.q[0] = *(const U4*)&s_p[wave * 16 + m][k0 + hi * 8];
    a.q[1] = *(const U4*)&s_p[wave * 16 + m][k0 + 16 + hi * 8];
#pragma unroll
    for (int nt = 0; nt < 4; ++nt) {
      FragAB b;
      const bf16* vrow = vtb + (size_t)(nt * 16 + m) * NREC + (size_t)g * 256 + k0 + hi * 8;
      b.q[0] = *(const U4*)(vrow);
      b.q[1] = *(const U4*)(vrow + 16);
      c2[nt].v = wmma_bf(a.v, b.v, c2[nt].v);
    }
  }
#pragma unroll
  for (int nt = 0; nt < 4; ++nt)
#pragma unroll
    for (int r = 0; r < 8; ++r)
      atn[((size_t)g * 48 + wave * 16 + r + hi * 8) * 64 + nt * 16 + m] = (bf16)c2[nt].f[r];
}

// ---------------- small VALU / prep kernels ----------------
__global__ void k_wt(const float* __restrict__ W, bf16* __restrict__ Wt, int K, int N, int Kpad) {
  int i = blockIdx.x * 256 + threadIdx.x;
  if (i >= N * Kpad) return;
  int n = i / Kpad, k = i % Kpad;
  Wt[i] = (k < K) ? (bf16)W[(size_t)k * N + n] : (bf16)0.f;
}
__global__ void k_w2rt(const float* __restrict__ e2w, const float* __restrict__ e2b, bf16* __restrict__ Wt) {
  int i = blockIdx.x * 256 + threadIdx.x;
  if (i >= 64 * 4160) return;
  int o = i / 4160, r = i % 4160, a = r >> 6, b = r & 63;
  float v = (a < 64) ? e2w[(size_t)a * 4096 + b * 64 + o] : e2b[b * 64 + o];
  Wt[i] = (bf16)v;
}
__global__ void k_embed_n(const float* __restrict__ scal, const int* __restrict__ cat,
                          const float* __restrict__ tab, bf16* __restrict__ out, int N) {
  int i = blockIdx.x * 256 + threadIdx.x;
  if (i >= N * 32) return;
  int n = i >> 5, c = i & 31;
  float v = (c < 16) ? scal[n * 16 + c] : tab[cat[n] * 16 + (c - 16)];
  out[i] = (bf16)v;
}
__global__ void k_embed_e(const float* __restrict__ scal, const int* __restrict__ cat,
                          const float* __restrict__ tab, bf16* __restrict__ out, int E) {
  int i = blockIdx.x * 256 + threadIdx.x;
  if (i >= E * 32) return;
  int e = i >> 5, c = i & 31;
  float v = (c < 4) ? scal[e * 4 + c] : ((c < 20) ? tab[cat[e] * 16 + (c - 4)] : 0.f);
  out[i] = (bf16)v;
}
__global__ void k_xrelu(const float* __restrict__ x, const float* __restrict__ b,
                        bf16* __restrict__ xb, int N) {
  int i = blockIdx.x * 256 + threadIdx.x;
  if (i >= N * 64) return;
  xb[i] = (bf16)fmaxf(x[i] + b[i & 63], 0.f);
}
__global__ void k_gru(const float* __restrict__ gi, const float* __restrict__ gh,
                      float* __restrict__ h, bf16* __restrict__ hb, int N) {
  int i = blockIdx.x * 256 + threadIdx.x;
  if (i >= N * 64) return;
  int n = i >> 6, f = i & 63;
  const float* gir = gi + (size_t)n * 192;
  const float* ghr = gh + (size_t)n * 192;
  float r = 1.f / (1.f + __expf(-(gir[f] + ghr[f])));
  float z = 1.f / (1.f + __expf(-(gir[64 + f] + ghr[64 + f])));
  float nn = tanhf(gir[128 + f] + r * ghr[128 + f]);
  float hv = h[i];
  hv = (1.f - z) * nn + z * hv;
  h[i] = hv;
  hb[i] = (bf16)hv;
}
__global__ void k_ln(const float* __restrict__ in, const float* __restrict__ g,
                     const float* __restrict__ be, float* __restrict__ outF,
                     bf16* __restrict__ outB, int rows, int cols) {
  int r = blockIdx.x * blockDim.x + threadIdx.x;
  if (r >= rows) return;
  const float* x = in + (size_t)r * cols;
  float mean = 0.f;
  for (int c = 0; c < cols; ++c) mean += x[c];
  mean /= cols;
  float var = 0.f;
  for (int c = 0; c < cols; ++c) { float d = x[c] - mean; var += d * d; }
  var /= cols;
  float inv = rsqrtf(var + 1e-5f);
  for (int c = 0; c < cols; ++c) {
    float v = (x[c] - mean) * inv * g[c] + be[c];
    v = (v > 0.f) ? v : 0.01f * v;
    outF[(size_t)r * cols + c] = v;
    if (outB) outB[(size_t)r * cols + c] = (bf16)v;
  }
}
__global__ void k_pack(const bf16* __restrict__ src, bf16* __restrict__ dst, int n) {
  int i = blockIdx.x * 256 + threadIdx.x;
  if (i >= n) return;
  dst[(size_t)(i >> 6) * 128 + (i & 63)] = src[i];
}
__global__ void k_nw(const float* __restrict__ h, const float* __restrict__ w,
                     const float* __restrict__ b, float* __restrict__ wn, int N) {
  int n = blockIdx.x * 256 + threadIdx.x;
  if (n >= N) return;
  float a = b[0];
  for (int f = 0; f < 64; ++f) a += h[(size_t)n * 64 + f] * w[f];
  wn[n] = 1.f / (1.f + __expf(-a));
}
__global__ void k_read(const float* __restrict__ h, const float* __restrict__ wn,
                       float* __restrict__ xcat, int npg, int colWS, int colMX) {
  int g = blockIdx.x, f = threadIdx.x;
  if (f >= 64) return;
  int basen = g * npg;
  float s = 0.f, mx = -3.4e38f;
  for (int i = 0; i < npg; ++i) {
    float hv = h[(size_t)(basen + i) * 64 + f];
    s += wn[basen + i] * hv;
    mx = fmaxf(mx, hv);
  }
  xcat[g * 256 + colWS + f] = s;
  xcat[g * 256 + colMX + f] = mx;
}
__global__ void k_f2b(const float* __restrict__ a, bf16* __restrict__ o, int n) {
  int i = blockIdx.x * 256 + threadIdx.x;
  if (i < n) o[i] = (bf16)a[i];
}
__global__ void k_out(const float* __restrict__ x, const float* __restrict__ w,
                      const float* __restrict__ b, float* __restrict__ out) {
  int g = threadIdx.x;
  if (g >= 16) return;
  float a = b[0];
  for (int f = 0; f < 64; ++f) a += x[g * 64 + f] * w[f];
  out[g] = a;
}

// ============================== host driver ==============================
extern "C" void kernel_launch(void* const* d_in, const int* in_sizes, int n_in,
                              void* d_out, int out_size, void* d_ws, size_t ws_size,
                              hipStream_t stream) {
  (void)in_sizes; (void)n_in; (void)out_size; (void)ws_size;
  auto F = [&](int i) { return (const float*)d_in[i]; };
  auto I = [&](int i) { return (const int*)d_in[i]; };

  const float* lig_scal = F(0);  const int* lig_cat = I(1);
  const float* rec_scal = F(2);  const int* rec_cat = I(3);
  const float* lig_escal = F(4); const int* lig_ecat = I(5);
  const float* rec_escal = F(6); const int* rec_ecat = I(7);
  const int* lig_src = I(8);  const int* lig_dst = I(9);
  const int* rec_src = I(10); const int* rec_dst = I(11);

  // param indices (setup_inputs insertion order, recursively flattened)
  const int GL0 = 18, GL1 = 29, GR0 = 40, GR1 = 51;   // gnn: +0 proj_w +1 proj_b +2 e1_w +3 e1_b +4 e2_w +5 e2_b +6 conv_b +7 w_ih +8 b_ih +9 w_hh +10 b_hh
  const int AT0 = 62, AT1 = 70, CB0 = 78, CB1 = 82;
  const int LR = 86, RR = 88, ML0 = 90, ML1 = 94, OW = 98, OB = 99;

  char* wsbase = (char*)d_ws;
  size_t off = 0;
  auto alloc = [&](size_t bytes) -> void* {
    off = (off + 255) & ~(size_t)255;
    void* p = wsbase + off;
    off += bytes;
    return p;
  };

  bf16* h0lig = (bf16*)alloc((size_t)NLIG * 32 * 2);
  bf16* h0rec = (bf16*)alloc((size_t)NREC * 32 * 2);
  bf16* eLig  = (bf16*)alloc((size_t)ELIG * 32 * 2);
  bf16* eRec  = (bf16*)alloc((size_t)EREC * 32 * 2);
  float* ligHf[2]; bf16* ligHb[2]; float* recHf[2]; bf16* recHb[2];
  for (int i = 0; i < 2; ++i) {
    ligHf[i] = (float*)alloc((size_t)NLIG * 64 * 4);
    ligHb[i] = (bf16*)alloc((size_t)NLIG * 64 * 2);
    recHf[i] = (float*)alloc((size_t)NREC * 64 * 4);
    recHb[i] = (bf16*)alloc((size_t)NREC * 64 * 2);
  }
  bf16* ehLig = (bf16*)alloc((size_t)ELIG * 64 * 2);
  bf16* ehRec = (bf16*)alloc((size_t)EREC * 64 * 2);
  float* xbuf = (float*)alloc((size_t)NREC * 64 * 4);
  bf16*  xbf  = (bf16*)alloc((size_t)NREC * 64 * 2);
  float* gi = (float*)alloc((size_t)NREC * 192 * 4);
  float* gh = (float*)alloc((size_t)NREC * 192 * 4);
  bf16 *projT[4], *e1T[4], *wihT[4], *whhT[4], *w2rT[4];
  for (int s = 0; s < 4; ++s) {
    projT[s] = (bf16*)alloc(64 * 64 * 2);
    e1T[s]   = (bf16*)alloc(64 * 32 * 2);
    wihT[s]  = (bf16*)alloc(192 * 64 * 2);
    whhT[s]  = (bf16*)alloc(192 * 64 * 2);
    w2rT[s]  = (bf16*)alloc((size_t)64 * 4160 * 2);
  }
  bf16 *wqT[2], *wkT[2], *wvT[2], *woT[2], *combT[2];
  for (int s = 0; s < 2; ++s) {
    wqT[s] = (bf16*)alloc(64 * 64 * 2);
    wkT[s] = (bf16*)alloc(64 * 64 * 2);
    wvT[s] = (bf16*)alloc(64 * 64 * 2);
    woT[s] = (bf16*)alloc(64 * 64 * 2);
    combT[s] = (bf16*)alloc(64 * 128 * 2);
  }
  bf16* mlpT0 = (bf16*)alloc(128 * 256 * 2);
  bf16* mlpT1 = (bf16*)alloc(64 * 128 * 2);
  bf16* qb   = (bf16*)alloc((size_t)NLIG * 64 * 2);
  bf16* kb   = (bf16*)alloc((size_t)NREC * 64 * 2);
  bf16* vtb  = (bf16*)alloc((size_t)64 * NREC * 2);
  bf16* atnb = (bf16*)alloc((size_t)NLIG * 64 * 2);
  bf16* cat2 = (bf16*)alloc((size_t)NLIG * 128 * 2);
  float* combF = (float*)alloc((size_t)NLIG * 64 * 4);
  float* wn   = (float*)alloc((size_t)NREC * 4);
  float* xcat = (float*)alloc(16 * 256 * 4);
  bf16*  xcatb = (bf16*)alloc(16 * 256 * 2);
  float* m1F = (float*)alloc(16 * 128 * 4);
  float* m1L = (float*)alloc(16 * 128 * 4);
  bf16*  m1b = (bf16*)alloc(16 * 128 * 2);
  float* m2F = (float*)alloc(16 * 64 * 4);
  float* m2L = (float*)alloc(16 * 64 * 4);

  auto wt = [&](const float* W, bf16* dst, int K, int N, int Kpad) {
    int tot = N * Kpad;
    k_wt<<<(tot + 255) / 256, 256, 0, stream>>>(W, dst, K, N, Kpad);
  };
  auto gemm = [&](const bf16* A, int lda, int K, const bf16* Wt, const float* bias,
                  float* oF, int ldF, bf16* oB, int ldB, int oT,
                  int M, int N, int act, float scale) {
    dim3 g(N / 16, (M + 63) / 64);
    k_gemm<<<g, 128, 0, stream>>>(A, lda, Wt, K, bias, oF, ldF, oB, ldB, oT, M, N, act, scale);
  };

  // ---- weight prep (bf16 + transpose; W2r rearrangement with bias folded) ----
  int gset[4] = {GL0, GL1, GR0, GR1};
  int projK[4] = {32, 64, 32, 64};
  for (int s = 0; s < 4; ++s) {
    wt(F(gset[s] + 0), projT[s], projK[s], 64, projK[s]);
    wt(F(gset[s] + 2), e1T[s], 20, 64, 32);
    wt(F(gset[s] + 7), wihT[s], 64, 192, 64);
    wt(F(gset[s] + 9), whhT[s], 64, 192, 64);
    k_w2rt<<<(64 * 4160 + 255) / 256, 256, 0, stream>>>(F(gset[s] + 4), F(gset[s] + 5), w2rT[s]);
  }
  for (int s = 0; s < 2; ++s) {
    int a = s ? AT1 : AT0, c = s ? CB1 : CB0;
    wt(F(a + 0), wqT[s], 64, 64, 64);
    wt(F(a + 2), wkT[s], 64, 64, 64);
    wt(F(a + 4), wvT[s], 64, 64, 64);
    wt(F(a + 6), woT[s], 64, 64, 64);
    wt(F(c + 0), combT[s], 128, 64, 128);
  }
  wt(F(ML0 + 0), mlpT0, 256, 128, 256);
  wt(F(ML1 + 0), mlpT1, 128, 64, 128);

  // ---- input embeddings (concat scal + table, pad to K=32) ----
  k_embed_n<<<(NLIG * 32 + 255) / 256, 256, 0, stream>>>(lig_scal, lig_cat, F(14), h0lig, NLIG);
  k_embed_n<<<(NREC * 32 + 255) / 256, 256, 0, stream>>>(rec_scal, rec_cat, F(15), h0rec, NREC);
  k_embed_e<<<(ELIG * 32 + 255) / 256, 256, 0, stream>>>(lig_escal, lig_ecat, F(16), eLig, ELIG);
  k_embed_e<<<(EREC * 32 + 255) / 256, 256, 0, stream>>>(rec_escal, rec_ecat, F(17), eRec, EREC);

  auto mpnn = [&](const bf16* Ain, int K, int Nn, int E,
                  const bf16* eFeat, const int* src, const int* dst,
                  int pb, int widx, bf16* ehBuf, float* hF, bf16* hB) {
    gemm(Ain, K, K, projT[widx], F(pb + 1), hF, 64, hB, 64, 0, Nn, 64, 1, 1.f);          // proj + relu
    gemm(eFeat, 32, 32, e1T[widx], F(pb + 3), nullptr, 0, ehBuf, 64, 0, E, 64, 1, 1.f);  // edge hidden + relu
    for (int it = 0; it < 3; ++it) {
      hipMemsetAsync(xbuf, 0, (size_t)Nn * 64 * 4, stream);
      k_edge_msg<<<E / 64, 128, 0, stream>>>(ehBuf, hB, src, dst, w2rT[widx], xbuf);
      k_xrelu<<<(Nn * 64 + 255) / 256, 256, 0, stream>>>(xbuf, F(pb + 6), xbf, Nn);
      gemm(xbf, 64, 64, wihT[widx], F(pb + 8), gi, 192, nullptr, 0, 0, Nn, 192, 0, 1.f);
      gemm(hB, 64, 64, whhT[widx], F(pb + 10), gh, 192, nullptr, 0, 0, Nn, 192, 0, 1.f);
      k_gru<<<(Nn * 64 + 255) / 256, 256, 0, stream>>>(gi, gh, hF, hB, Nn);
    }
  };

  auto attn = [&](int s, const bf16* ligb, const bf16* recb, float* outF, bf16* outB) {
    int a = s ? AT1 : AT0, c = s ? CB1 : CB0;
    gemm(ligb, 64, 64, wqT[s], F(a + 1), nullptr, 0, qb, 64, 0, NLIG, 64, 0, 0.125f);    // q, scaled 1/sqrt(H)
    gemm(recb, 64, 64, wkT[s], F(a + 3), nullptr, 0, kb, 64, 0, NREC, 64, 0, 1.f);
    gemm(recb, 64, 64, wvT[s], F(a + 5), nullptr, 0, vtb, NREC, 1, NREC, 64, 0, 1.f);    // v, transposed store
    k_attn<<<16, 96, 0, stream>>>(qb, kb, vtb, atnb);
    k_pack<<<(NLIG * 64 + 255) / 256, 256, 0, stream>>>(ligb, cat2, NLIG * 64);          // cat2[:, :64] = lig_h
    gemm(atnb, 64, 64, woT[s], F(a + 7), nullptr, 0, cat2 + 64, 128, 0, NLIG, 64, 0, 1.f); // cat2[:, 64:]
    gemm(cat2, 128, 128, combT[s], F(c + 1), combF, 64, nullptr, 0, 0, NLIG, 64, 0, 1.f);
    k_ln<<<(NLIG + 127) / 128, 128, 0, stream>>>(combF, F(c + 2), F(c + 3), outF, outB, NLIG, 64);
  };

  // ---- stage 0 ----
  mpnn(h0rec, 32, NREC, EREC, eRec, rec_src, rec_dst, GR0, 2, ehRec, recHf[0], recHb[0]);
  mpnn(h0lig, 32, NLIG, ELIG, eLig, lig_src, lig_dst, GL0, 0, ehLig, ligHf[0], ligHb[0]);
  attn(0, ligHb[0], recHb[0], ligHf[1], ligHb[1]);
  // ---- stage 1 ----
  mpnn(recHb[0], 64, NREC, EREC, eRec, rec_src, rec_dst, GR1, 3, ehRec, recHf[1], recHb[1]);
  mpnn(ligHb[1], 64, NLIG, ELIG, eLig, lig_src, lig_dst, GL1, 1, ehLig, ligHf[0], ligHb[0]);
  attn(1, ligHb[0], recHb[1], ligHf[1], ligHb[1]);

  // ---- readout + MLP head ----
  k_nw<<<(NREC + 255) / 256, 256, 0, stream>>>(recHf[1], F(RR + 0), F(RR + 1), wn, NREC);
  k_read<<<16, 64, 0, stream>>>(recHf[1], wn, xcat, 256, 0, 64);
  k_nw<<<(NLIG + 255) / 256, 256, 0, stream>>>(ligHf[1], F(LR + 0), F(LR + 1), wn, NLIG);
  k_read<<<16, 64, 0, stream>>>(ligHf[1], wn, xcat, 48, 128, 192);
  k_f2b<<<(16 * 256 + 255) / 256, 256, 0, stream>>>(xcat, xcatb, 16 * 256);
  gemm(xcatb, 256, 256, mlpT0, F(ML0 + 1), m1F, 128, nullptr, 0, 0, 16, 128, 0, 1.f);
  k_ln<<<1, 64, 0, stream>>>(m1F, F(ML0 + 2), F(ML0 + 3), m1L, m1b, 16, 128);
  gemm(m1b, 128, 128, mlpT1, F(ML1 + 1), m2F, 64, nullptr, 0, 0, 16, 64, 0, 1.f);
  k_ln<<<1, 64, 0, stream>>>(m2F, F(ML1 + 2), F(ML1 + 3), m2L, nullptr, 16, 64);
  k_out<<<1, 16, 0, stream>>>(m2L, F(OW), F(OB), (float*)d_out);
}
